// ExpertChoiceFFN_11441792877183
// MI455X (gfx1250) — matrix operand order repaired and checked by reading, not compiled, not run on hardware
//
#include <hip/hip_runtime.h>
#include <hip/hip_bf16.h>

// ---------------------------------------------------------------------------
// ExpertChoiceFFN for MI455X (gfx1250, wave32, WMMA + TDM async tensor loads)
// D=1024, F=4096, E=8, N=8192, CAP=2048
// ---------------------------------------------------------------------------

#define DMODEL 1024
#define DFF    4096
#define NEXP   8
#define NTOK   8192
#define CAP    2048

typedef __attribute__((ext_vector_type(16))) __bf16 v16bf;
typedef __attribute__((ext_vector_type(8)))  float  v8f;
typedef __attribute__((ext_vector_type(4)))  unsigned int u32x4;
typedef __attribute__((ext_vector_type(8)))  int          i32x8;
typedef __attribute__((ext_vector_type(4)))  int          i32x4;

union FragU { v16bf v; uint4 q[2]; };

static __device__ __forceinline__ unsigned short f2bf(float f) {
    unsigned int u = __float_as_uint(f);
    unsigned int r = u + 0x7FFFu + ((u >> 16) & 1u);   // RNE
    return (unsigned short)(r >> 16);
}

static __device__ __forceinline__ v8f wmma_bf16(const v16bf& a, const v16bf& b, v8f c) {
    return __builtin_amdgcn_wmma_f32_16x16x32_bf16(false, a, false, b, (short)0, c,
                                                   false, false);
}

// ---------------------------------------------------------------------------
// Tensor Data Mover: async 2D bf16 tile load (global -> LDS).
// This toolchain exposes the 6-arg builtin:
//   (u32x4 g0, i32x8 g1, i32x4 g2, i32x4 g3, i32x8 extra, i32 cpol)
// Descriptor per CDNA5 ISA:
//   group0: count=1 | lds_addr | global_addr[56:0] | type=2
//   group1: data_size=2B, tensor_dim0/1 (huge, no OOB), tile_dim0 (row elems),
//           tile_dim1 (rows), tensor_dim0_stride (row stride in elems)
//   groups 2/3: zero (2D tile, tile_dim2 = 0 -> unused)
// ---------------------------------------------------------------------------
static __device__ __forceinline__ void tdm_load_2d_bf16(unsigned lds_addr,
                                                        const void* gptr,
                                                        unsigned tile_w_elems,
                                                        unsigned tile_h_rows,
                                                        unsigned long long stride_elems) {
    unsigned long long ga = (unsigned long long)(size_t)gptr;
    u32x4 g0;
    g0[0] = 1u;                                           // count=1, user mode
    g0[1] = lds_addr;                                     // LDS byte address
    g0[2] = (unsigned)(ga & 0xFFFFFFFFu);                 // global addr [95:64]
    g0[3] = (unsigned)((ga >> 32) & 0x01FFFFFFu) | (2u << 30); // [120:96] | type=2

    const unsigned TD0 = 0x40000000u;                     // tensor_dim0 (huge)
    const unsigned TD1 = 0x40000000u;                     // tensor_dim1 (huge)
    i32x8 g1;
    g1[0] = (int)(1u << 16);                              // data_size = 1 (2 bytes)
    g1[1] = (int)((TD0 & 0xFFFFu) << 16);                 // tensor_dim0[15:0] @ 63:48
    g1[2] = (int)((TD0 >> 16) | ((TD1 & 0xFFFFu) << 16)); // td0[31:16] | td1[15:0]
    g1[3] = (int)((TD1 >> 16) | ((tile_w_elems & 0xFFFFu) << 16)); // td1 hi | tile_dim0
    g1[4] = (int)(tile_h_rows & 0xFFFFu);                 // tile_dim1, tile_dim2=0
    g1[5] = (int)(stride_elems & 0xFFFFFFFFull);          // dim0_stride[31:0]
    g1[6] = (int)((stride_elems >> 32) & 0xFFFFull);      // dim0_stride[47:32]
    g1[7] = 0;

    i32x4 gz4 = {0, 0, 0, 0};
    i32x8 gz8 = {0, 0, 0, 0, 0, 0, 0, 0};
    __builtin_amdgcn_tensor_load_to_lds(g0, g1, gz4, gz4, gz8, 0);
}

// A fragment: 16x32 bf16 tile, LDS row-major [rows][32]
static __device__ __forceinline__ v16bf load_a_frag(const unsigned short* As, int rowBase, int lane) {
    int lm = lane & 15, hi = lane >> 4;
    FragU u;
    u.q[0] = *(const uint4*)(As + (rowBase + lm) * 32 + hi * 8);
    u.q[1] = *(const uint4*)(As + (rowBase + lm) * 32 + 16 + hi * 8);
    return u.v;
}

// B fragment: 32x16 (KxN) bf16 tile staged N-major in LDS [cols][32]
static __device__ __forceinline__ v16bf load_b_frag(const unsigned short* Bt, int colBase, int lane) {
    int lm = lane & 15, hi = lane >> 4;
    const unsigned short* p = Bt + (colBase + lm) * 32 + hi * 8;
    FragU u;
    u.q[0] = *(const uint4*)(p);
    u.q[1] = *(const uint4*)(p + 16);
    return u.v;
}

// ---------------------------------------------------------------------------
// 1) Router: scores[e][n] = dot(x[n], Wr[e])
// ---------------------------------------------------------------------------
__global__ __launch_bounds__(256) void router_kernel(const float* __restrict__ x,
                                                     const float* __restrict__ Wr,
                                                     float* __restrict__ scores) {
    int n = blockIdx.x * blockDim.x + threadIdx.x;
    if (n >= NTOK) return;
    const float4* xv = (const float4*)(x + (size_t)n * DMODEL);
    const float4* wr = (const float4*)Wr;
    float acc[NEXP];
#pragma unroll
    for (int e = 0; e < NEXP; ++e) acc[e] = 0.f;
    for (int k4 = 0; k4 < DMODEL / 4; ++k4) {
        float4 xx = xv[k4];
#pragma unroll
        for (int e = 0; e < NEXP; ++e) {
            float4 ww = wr[e * (DMODEL / 4) + k4];
            acc[e] += xx.x * ww.x + xx.y * ww.y + xx.z * ww.z + xx.w * ww.w;
        }
    }
#pragma unroll
    for (int e = 0; e < NEXP; ++e) scores[(size_t)e * NTOK + n] = acc[e];
}

// ---------------------------------------------------------------------------
// 2) Per-expert top-CAP selection (bitonic sort in LDS) + softmax weights
// ---------------------------------------------------------------------------
__global__ __launch_bounds__(1024) void topk_kernel(const float* __restrict__ scores,
                                                    int* __restrict__ top_idx,
                                                    float* __restrict__ wts) {
    __shared__ unsigned long long arr[NTOK];   // 64 KB of 320 KB/WGP LDS
    const int e = blockIdx.x;
    const int tid = threadIdx.x;

    for (int i = tid; i < NTOK; i += 1024) {
        unsigned int b = __float_as_uint(scores[(size_t)e * NTOK + i]);
        unsigned int key = (b & 0x80000000u) ? ~b : (b | 0x80000000u);  // orderable
        arr[i] = ((unsigned long long)key << 32) | (unsigned int)i;
    }
    __syncthreads();

    for (int k = 2; k <= NTOK; k <<= 1) {
        for (int j = k >> 1; j > 0; j >>= 1) {
            for (int i = tid; i < NTOK; i += 1024) {
                int ixj = i ^ j;
                if (ixj > i) {
                    unsigned long long a = arr[i], b = arr[ixj];
                    bool up = ((i & k) == 0);
                    if (up ? (a < b) : (a > b)) { arr[i] = b; arr[ixj] = a; }
                }
            }
            __syncthreads();
        }
    }

    auto dec = [](unsigned long long p) -> float {
        unsigned int u = (unsigned int)(p >> 32);
        unsigned int b = (u & 0x80000000u) ? (u ^ 0x80000000u) : ~u;
        return __uint_as_float(b);
    };

    float vmax = dec(arr[0]);
    float local = 0.f;
    for (int c = tid; c < CAP; c += 1024) local += __expf(dec(arr[c]) - vmax);

    float* scratch = (float*)(arr + NTOK / 2);   // overlays unused tail
#pragma unroll
    for (int o = 16; o > 0; o >>= 1) local += __shfl_xor(local, o, 32);
    if ((tid & 31) == 0) scratch[tid >> 5] = local;
    __syncthreads();
    if (tid < 32) {
        float v = scratch[tid];
#pragma unroll
        for (int o = 16; o > 0; o >>= 1) v += __shfl_xor(v, o, 32);
        if (tid == 0) scratch[0] = v;
    }
    __syncthreads();
    float inv_sum = 1.f / scratch[0];

    for (int c = tid; c < CAP; c += 1024) {
        unsigned long long p = arr[c];
        top_idx[(size_t)e * CAP + c] = (int)(unsigned int)(p & 0xFFFFFFFFu);
        wts[(size_t)e * CAP + c] = __expf(dec(p) - vmax) * inv_sum;
    }
}

// ---------------------------------------------------------------------------
// 3) Gather chosen tokens -> bf16  Xe[e][c][d]
// ---------------------------------------------------------------------------
__global__ __launch_bounds__(256) void gather_kernel(const float* __restrict__ x,
                                                     const int* __restrict__ top_idx,
                                                     unsigned short* __restrict__ Xe) {
    int row = blockIdx.x;              // e*CAP + c
    int tok = top_idx[row];
    const float4* src = (const float4*)(x + (size_t)tok * DMODEL);
    ushort4* dst = (ushort4*)(Xe + (size_t)row * DMODEL);
    int t = threadIdx.x;
    float4 v = src[t];
    ushort4 o;
    o.x = f2bf(v.x); o.y = f2bf(v.y); o.z = f2bf(v.z); o.w = f2bf(v.w);
    dst[t] = o;
}

// ---------------------------------------------------------------------------
// 4) Fused gate/up GEMM + SiLU.  A tiles via TDM (double-buffered, TENSORcnt),
//    B tiles via fp32->bf16 convert, packed ds_store_b32.
// ---------------------------------------------------------------------------
__global__ __launch_bounds__(256) void gateup_kernel(const unsigned short* __restrict__ Xe,
                                                     const float* __restrict__ Wg,
                                                     const float* __restrict__ Wu,
                                                     unsigned short* __restrict__ hid) {
    __shared__ unsigned short As[2][128 * 32];
    __shared__ unsigned short BgT[64 * 32];
    __shared__ unsigned short BuT[64 * 32];

    const int e = blockIdx.z;
    const int nBlk = blockIdx.x * 64;
    const int mBlk = blockIdx.y * 128;
    const int tid = threadIdx.x;
    const int lane = tid & 31;
    const int w = tid >> 5;
    const int wm = (w & 3) * 32;
    const int wn = (w >> 2) * 32;

    const unsigned ldsA[2] = { (unsigned)(size_t)&As[0][0], (unsigned)(size_t)&As[1][0] };
    const unsigned short* Abase = Xe + (size_t)(e * CAP + mBlk) * DMODEL;

    const v8f zero = {0.f, 0.f, 0.f, 0.f, 0.f, 0.f, 0.f, 0.f};
    v8f G[2][2], U[2][2];
#pragma unroll
    for (int i = 0; i < 2; ++i)
#pragma unroll
        for (int j = 0; j < 2; ++j) { G[i][j] = zero; U[i][j] = zero; }

    if (w == 0) tdm_load_2d_bf16(ldsA[0], Abase, 32, 128, DMODEL);  // prologue tile

    int buf = 0;
    for (int k0 = 0; k0 < DMODEL; k0 += 32) {
        __syncthreads();                      // prev compute done; B LDS reusable

        // ---- stage B (32x64 fp32 -> bf16, N-major, packed k-pairs) ----
        {
            int k  = (tid >> 4) * 2;          // even k
            int c4 = (tid & 15) * 4;
            size_t g0off = ((size_t)e * DMODEL + (size_t)(k0 + k)) * DFF + nBlk + c4;
            float4 ga = *(const float4*)(Wg + g0off);
            float4 gb = *(const float4*)(Wg + g0off + DFF);
            float4 ua = *(const float4*)(Wu + g0off);
            float4 ub = *(const float4*)(Wu + g0off + DFF);
            if (k0 + 32 < DMODEL) {
                __builtin_prefetch(Wg + g0off + (size_t)32 * DFF, 0, 1);
                __builtin_prefetch(Wu + g0off + (size_t)32 * DFF, 0, 1);
            }
            unsigned pg0 = f2bf(ga.x) | ((unsigned)f2bf(gb.x) << 16);
            unsigned pg1 = f2bf(ga.y) | ((unsigned)f2bf(gb.y) << 16);
            unsigned pg2 = f2bf(ga.z) | ((unsigned)f2bf(gb.z) << 16);
            unsigned pg3 = f2bf(ga.w) | ((unsigned)f2bf(gb.w) << 16);
            unsigned pu0 = f2bf(ua.x) | ((unsigned)f2bf(ub.x) << 16);
            unsigned pu1 = f2bf(ua.y) | ((unsigned)f2bf(ub.y) << 16);
            unsigned pu2 = f2bf(ua.z) | ((unsigned)f2bf(ub.z) << 16);
            unsigned pu3 = f2bf(ua.w) | ((unsigned)f2bf(ub.w) << 16);
            *(unsigned*)(BgT + (c4 + 0) * 32 + k) = pg0;
            *(unsigned*)(BgT + (c4 + 1) * 32 + k) = pg1;
            *(unsigned*)(BgT + (c4 + 2) * 32 + k) = pg2;
            *(unsigned*)(BgT + (c4 + 3) * 32 + k) = pg3;
            *(unsigned*)(BuT + (c4 + 0) * 32 + k) = pu0;
            *(unsigned*)(BuT + (c4 + 1) * 32 + k) = pu1;
            *(unsigned*)(BuT + (c4 + 2) * 32 + k) = pu2;
            *(unsigned*)(BuT + (c4 + 3) * 32 + k) = pu3;
        }

        // ---- TDM: queue next A tile, guarantee current tile resident ----
        if (w == 0) {
            if (k0 + 32 < DMODEL) {
                tdm_load_2d_bf16(ldsA[buf ^ 1], Abase + k0 + 32, 32, 128, DMODEL);
                __builtin_amdgcn_s_wait_tensorcnt(1);   // in-order: tile k0 done
            } else {
                __builtin_amdgcn_s_wait_tensorcnt(0);
            }
        }
        __syncthreads();

        const unsigned short* Ab = &As[buf][0];
        v16bf a0 = load_a_frag(Ab, wm + 0, lane);
        v16bf a1 = load_a_frag(Ab, wm + 16, lane);
        v16bf bg0 = load_b_frag(BgT, wn + 0, lane);
        v16bf bg1 = load_b_frag(BgT, wn + 16, lane);
        v16bf bu0 = load_b_frag(BuT, wn + 0, lane);
        v16bf bu1 = load_b_frag(BuT, wn + 16, lane);

        G[0][0] = wmma_bf16(a0, bg0, G[0][0]);
        G[0][1] = wmma_bf16(a0, bg1, G[0][1]);
        G[1][0] = wmma_bf16(a1, bg0, G[1][0]);
        G[1][1] = wmma_bf16(a1, bg1, G[1][1]);
        U[0][0] = wmma_bf16(a0, bu0, U[0][0]);
        U[0][1] = wmma_bf16(a0, bu1, U[0][1]);
        U[1][0] = wmma_bf16(a1, bu0, U[1][0]);
        U[1][1] = wmma_bf16(a1, bu1, U[1][1]);

        buf ^= 1;
    }

    // epilogue: silu(gate)*up -> bf16 hid
    const int lm = lane & 15, hi = lane >> 4;
#pragma unroll
    for (int i = 0; i < 2; ++i)
#pragma unroll
        for (int j = 0; j < 2; ++j) {
            v8f g = G[i][j], u = U[i][j];
#pragma unroll
            for (int r = 0; r < 8; ++r) {
                float gv = g[r], uv = u[r];
                float h = __fdividef(gv, 1.f + __expf(-gv)) * uv;
                int m = mBlk + wm + i * 16 + r + hi * 8;
                int n = nBlk + wn + j * 16 + lm;
                hid[((size_t)(e * CAP) + m) * DFF + n] = f2bf(h);
            }
        }
}

// ---------------------------------------------------------------------------
// 5) Down GEMM + weighted scatter-add.  A tiles (hid, bf16) via TDM.
// ---------------------------------------------------------------------------
__global__ __launch_bounds__(256) void down_kernel(const unsigned short* __restrict__ hid,
                                                   const float* __restrict__ Wd,
                                                   const int* __restrict__ top_idx,
                                                   const float* __restrict__ wts,
                                                   float* __restrict__ out) {
    __shared__ unsigned short As[2][128 * 32];
    __shared__ unsigned short BT[64 * 32];

    const int e = blockIdx.z;
    const int nBlk = blockIdx.x * 64;
    const int mBlk = blockIdx.y * 128;
    const int tid = threadIdx.x;
    const int lane = tid & 31;
    const int w = tid >> 5;
    const int wm = (w & 3) * 32;
    const int wn = (w >> 2) * 32;

    const unsigned ldsA[2] = { (unsigned)(size_t)&As[0][0], (unsigned)(size_t)&As[1][0] };
    const unsigned short* Abase = hid + (size_t)(e * CAP + mBlk) * DFF;

    const v8f zero = {0.f, 0.f, 0.f, 0.f, 0.f, 0.f, 0.f, 0.f};
    v8f Acc[2][2];
#pragma unroll
    for (int i = 0; i < 2; ++i)
#pragma unroll
        for (int j = 0; j < 2; ++j) Acc[i][j] = zero;

    if (w == 0) tdm_load_2d_bf16(ldsA[0], Abase, 32, 128, DFF);

    int buf = 0;
    for (int k0 = 0; k0 < DFF; k0 += 32) {
        __syncthreads();

        {
            int k  = (tid >> 4) * 2;
            int c4 = (tid & 15) * 4;
            size_t g0off = ((size_t)e * DFF + (size_t)(k0 + k)) * DMODEL + nBlk + c4;
            float4 ba = *(const float4*)(Wd + g0off);
            float4 bb = *(const float4*)(Wd + g0off + DMODEL);
            if (k0 + 32 < DFF) {
                __builtin_prefetch(Wd + g0off + (size_t)32 * DMODEL, 0, 1);
            }
            unsigned p0 = f2bf(ba.x) | ((unsigned)f2bf(bb.x) << 16);
            unsigned p1 = f2bf(ba.y) | ((unsigned)f2bf(bb.y) << 16);
            unsigned p2 = f2bf(ba.z) | ((unsigned)f2bf(bb.z) << 16);
            unsigned p3 = f2bf(ba.w) | ((unsigned)f2bf(bb.w) << 16);
            *(unsigned*)(BT + (c4 + 0) * 32 + k) = p0;
            *(unsigned*)(BT + (c4 + 1) * 32 + k) = p1;
            *(unsigned*)(BT + (c4 + 2) * 32 + k) = p2;
            *(unsigned*)(BT + (c4 + 3) * 32 + k) = p3;
        }

        if (w == 0) {
            if (k0 + 32 < DFF) {
                tdm_load_2d_bf16(ldsA[buf ^ 1], Abase + k0 + 32, 32, 128, DFF);
                __builtin_amdgcn_s_wait_tensorcnt(1);
            } else {
                __builtin_amdgcn_s_wait_tensorcnt(0);
            }
        }
        __syncthreads();

        const unsigned short* Ab = &As[buf][0];
        v16bf a0 = load_a_frag(Ab, wm + 0, lane);
        v16bf a1 = load_a_frag(Ab, wm + 16, lane);
        v16bf b0 = load_b_frag(BT, wn + 0, lane);
        v16bf b1 = load_b_frag(BT, wn + 16, lane);

        Acc[0][0] = wmma_bf16(a0, b0, Acc[0][0]);
        Acc[0][1] = wmma_bf16(a0, b1, Acc[0][1]);
        Acc[1][0] = wmma_bf16(a1, b0, Acc[1][0]);
        Acc[1][1] = wmma_bf16(a1, b1, Acc[1][1]);

        buf ^= 1;
    }

    const int lm = lane & 15, hi = lane >> 4;
#pragma unroll
    for (int i = 0; i < 2; ++i)
#pragma unroll
        for (int j = 0; j < 2; ++j) {
            v8f a = Acc[i][j];
#pragma unroll
            for (int r = 0; r < 8; ++r) {
                int m = mBlk + wm + i * 16 + r + hi * 8;
                int tok = top_idx[(size_t)e * CAP + m];
                float wgt = wts[(size_t)e * CAP + m];
                int n = nBlk + wn + j * 16 + lm;
                __hip_atomic_fetch_add(out + (size_t)tok * DMODEL + n, wgt * a[r],
                                       __ATOMIC_RELAXED, __HIP_MEMORY_SCOPE_AGENT);
            }
        }
}

// ---------------------------------------------------------------------------
// launch
// ---------------------------------------------------------------------------
extern "C" void kernel_launch(void* const* d_in, const int* in_sizes, int n_in,
                              void* d_out, int out_size, void* d_ws, size_t ws_size,
                              hipStream_t stream) {
    const float* x  = (const float*)d_in[0];
    const float* Wr = (const float*)d_in[1];
    const float* Wg = (const float*)d_in[2];
    const float* Wu = (const float*)d_in[3];
    const float* Wd = (const float*)d_in[4];
    float* out = (float*)d_out;

    char* ws = (char*)d_ws;
    size_t off = 0;
    float* scores = (float*)(ws + off);                off += (size_t)NEXP * NTOK * 4;
    int* top_idx  = (int*)(ws + off);                  off += (size_t)NEXP * CAP * 4;
    float* wts    = (float*)(ws + off);                off += (size_t)NEXP * CAP * 4;
    unsigned short* Xe  = (unsigned short*)(ws + off); off += (size_t)NEXP * CAP * DMODEL * 2;
    unsigned short* hid = (unsigned short*)(ws + off); off += (size_t)NEXP * CAP * DFF * 2;
    (void)ws_size;

    (void)hipMemsetAsync(d_out, 0, (size_t)out_size * sizeof(float), stream);

    router_kernel<<<NTOK / 256, 256, 0, stream>>>(x, Wr, scores);
    topk_kernel<<<NEXP, 1024, 0, stream>>>(scores, top_idx, wts);
    gather_kernel<<<NEXP * CAP, 256, 0, stream>>>(x, top_idx, Xe);
    gateup_kernel<<<dim3(DFF / 64, CAP / 128, NEXP), 256, 0, stream>>>(Xe, Wg, Wu, hid);
    down_kernel<<<dim3(DMODEL / 64, CAP / 128, NEXP), 256, 0, stream>>>(hid, Wd, top_idx, wts, out);
}
